// PerceiverAttentionLayer_25013889532501
// MI455X (gfx1250) — matrix operand order repaired and checked by reading, not compiled, not run on hardware
//
#include <hip/hip_runtime.h>
#include <hip/hip_bf16.h>
#include <cstdint>

// ---------------------------------------------------------------------------
// Perceiver attention layer for MI455X (gfx1250, wave32, WMMA bf16 path).
// All heavy matmuls use v_wmma_f32_16x16x32_bf16 (f32 accumulate).
// ---------------------------------------------------------------------------

typedef __bf16 bf16;
typedef __attribute__((ext_vector_type(16))) __bf16 v16bf;
typedef __attribute__((ext_vector_type(8)))  __bf16 v8bf;
typedef __attribute__((ext_vector_type(8)))  float  v8f;

#define DEV __device__ __forceinline__

constexpr int Bn    = 8;
constexpr int Fn    = 4096;
constexpr int Qc    = 64;
constexpr int Dn    = 1024;      // model dim == K of projections
constexpr int Hn    = 16;
constexpr int DHn   = 64;
constexpr int FQ    = Fn + Qc;   // 4160 keys
constexpr int INNER = Hn * DHn;  // 1024

// ---------------------------------------------------------------------------
// WMMA helpers (layouts per CDNA5 ISA 7.12.2, wave32)
// ---------------------------------------------------------------------------

DEV v8f wmma_bf16(v16bf a, v16bf b, v8f c) {
    return __builtin_amdgcn_wmma_f32_16x16x32_bf16(
        /*neg_a=*/false, a, /*neg_b=*/false, b,
        /*c_mod=*/(short)0, c, /*reuse_a=*/false, /*reuse_b=*/false);
}

// A fragment: 16x32 (MxK). Lane L: M = m0 + (L&15), half = L>>4.
// regs j=0..7  -> K = k0 + 8*half + j      (contiguous 16B)
// regs j=8..15 -> K = k0 + 16 + 8*half + (j-8) (contiguous 16B)
DEV v16bf frag_a(const bf16* __restrict__ A, int lda, int m0, int k0) {
    int lane = threadIdx.x & 31;
    int lh   = lane >> 4;
    int m    = m0 + (lane & 15);
    const bf16* p = A + (size_t)m * lda + k0 + 8 * lh;
    union { v16bf v; v8bf h[2]; } u;
    u.h[0] = *(const v8bf*)(p);
    u.h[1] = *(const v8bf*)(p + 16);
    return u.v;
}

// B fragment: 32x16 (KxN), source stored N-major (Bt[n][k]).
// Lane L: N = n0 + (L&15), half = L>>4; regs j=0..7 -> K = k0 + 8*half + j, j=8..15 -> K = k0 + 16 + 8*half + (j-8).
DEV v16bf frag_b(const bf16* __restrict__ Bt, int ldb, int n0, int k0) {
    int lane = threadIdx.x & 31;
    int lh   = lane >> 4;
    int n    = n0 + (lane & 15);
    const bf16* p = Bt + (size_t)n * ldb + k0 + 8 * lh;
    union { v16bf v; v8bf h[2]; } u;
    u.h[0] = *(const v8bf*)(p);
    u.h[1] = *(const v8bf*)(p + 16);
    return u.v;
}

// C/D layout: lane L -> N = (L&15); reg v -> M = v + 8*(L>>4)

// ---------------------------------------------------------------------------
// Kernel 1: weight convert f32 -> bf16 with transpose: Wt[n][k] = W[k][n]
// ---------------------------------------------------------------------------
__global__ __launch_bounds__(256)
void k_wconv(const float* __restrict__ W, bf16* __restrict__ Wt) {
    int idx = blockIdx.x * 256 + threadIdx.x;   // over 1024*1024
    int k = idx >> 10;
    int n = idx & 1023;
    Wt[(size_t)n * Dn + k] = (bf16)W[idx];
}

// ---------------------------------------------------------------------------
// Kernel 2: LayerNorm -> bf16.  One block per row (D = 1024, 256 threads x4).
// Features: out row b*FQ + f.  Latents: out rows b*FQ + F + q AND compact.
// ---------------------------------------------------------------------------
__global__ __launch_bounds__(256)
void k_layernorm(const float* __restrict__ X, const float* __restrict__ w,
                 const float* __restrict__ bias, bf16* __restrict__ out_kv,
                 bf16* __restrict__ out_lat, int rows_per_batch, int is_lat) {
    int r = blockIdx.x;
    int b = r / rows_per_batch;
    int rr = r - b * rows_per_batch;
    size_t kv_row = (size_t)b * FQ + (is_lat ? (Fn + rr) : rr);
    const float* x = X + (size_t)r * Dn;

    __shared__ float s1[256], s2[256];
    int t = threadIdx.x;
    float v[4];
    float s = 0.f, ss = 0.f;
#pragma unroll
    for (int i = 0; i < 4; ++i) {
        v[i] = x[t + 256 * i];
        s  += v[i];
        ss += v[i] * v[i];
    }
    s1[t] = s; s2[t] = ss;
    __syncthreads();
    for (int o = 128; o > 0; o >>= 1) {
        if (t < o) { s1[t] += s1[t + o]; s2[t] += s2[t + o]; }
        __syncthreads();
    }
    float mu  = s1[0] * (1.0f / Dn);
    float var = s2[0] * (1.0f / Dn) - mu * mu;
    float rstd = rsqrtf(var + 1e-5f);

    bf16* okv = out_kv + kv_row * Dn;
    bf16* ola = out_lat ? out_lat + (size_t)r * Dn : nullptr;
#pragma unroll
    for (int i = 0; i < 4; ++i) {
        int c = t + 256 * i;
        float y = (v[i] - mu) * rstd * w[c] + bias[c];
        okv[c] = (bf16)y;
        if (ola) ola[c] = (bf16)y;
    }
}

// ---------------------------------------------------------------------------
// Kernel 3: general bf16 WMMA GEMM.  C[M,N] = alpha * A[M,K] @ Bt[N,K]^T.
// Block tile 128x128 (8 waves: 4(m) x 2(n); wave tile 32x64).
// mode 0: bf16 row-major out; mode 1: per-head-transposed V store;
// mode 2: f32 row-major out (final projection to d_out).
// Requires M%128==0, N%128==0, K%32==0.
// ---------------------------------------------------------------------------
__global__ __launch_bounds__(256)
void k_gemm(const bf16* __restrict__ A, const bf16* __restrict__ Bt,
            void* __restrict__ Cout, int K, int N, float alpha, int mode) {
    int wave = threadIdx.x >> 5;
    int wm = wave & 3;
    int wn = wave >> 2;
    int m0 = blockIdx.x * 128 + wm * 32;
    int n0 = blockIdx.y * 128 + wn * 64;

    v8f acc[2][4] = {};
    for (int k0 = 0; k0 < K; k0 += 32) {
        v16bf a0 = frag_a(A, K, m0, k0);
        v16bf a1 = frag_a(A, K, m0 + 16, k0);
#pragma unroll
        for (int nf = 0; nf < 4; ++nf) {
            v16bf b = frag_b(Bt, K, n0 + nf * 16, k0);
            acc[0][nf] = wmma_bf16(a0, b, acc[0][nf]);
            acc[1][nf] = wmma_bf16(a1, b, acc[1][nf]);
        }
    }

    int lane = threadIdx.x & 31;
    int lh = lane >> 4;
    int nl = lane & 15;
#pragma unroll
    for (int mf = 0; mf < 2; ++mf) {
#pragma unroll
        for (int nf = 0; nf < 4; ++nf) {
            int n = n0 + nf * 16 + nl;
#pragma unroll
            for (int v = 0; v < 8; ++v) {
                int m = m0 + mf * 16 + 8 * lh + v;
                float val = acc[mf][nf][v] * alpha;
                if (mode == 0) {
                    ((bf16*)Cout)[(size_t)m * N + n] = (bf16)val;
                } else if (mode == 1) {
                    // row m = b*FQ + f ; col n = h*64 + dh  ->  Vt[b][h][dh][f]
                    int bb = m / FQ, f = m - bb * FQ;
                    int hh = n >> 6, dh = n & 63;
                    ((bf16*)Cout)[(((size_t)(bb * Hn + hh) * DHn + dh) * FQ) + f] = (bf16)val;
                } else {
                    ((float*)Cout)[(size_t)m * N + n] = val;
                }
            }
        }
    }
}

// ---------------------------------------------------------------------------
// Kernel 4: sim = (q*scale) @ K^T with mask.  grid (65 key-blocks, B*H).
// Each block: 64 queries x 64 keys; 8 waves: wave = (qt 0..3) x (2 key tiles).
// ---------------------------------------------------------------------------
__global__ __launch_bounds__(256)
void k_sim(const bf16* __restrict__ qb, const bf16* __restrict__ Kb,
           const int* __restrict__ mask, float* __restrict__ sim) {
    int bh = blockIdx.y;
    int b = bh >> 4, h = bh & 15;
    int fblk = blockIdx.x;                 // 0..64
    int wave = threadIdx.x >> 5;
    int qt = wave & 3;
    int ng = wave >> 2;

    int qrow0 = b * Qc + qt * 16;          // row in q buffer [512,1024]
    int col0 = h * DHn;                    // dh offset within inner

    v8f acc[2] = {};
    for (int k0 = 0; k0 < DHn; k0 += 32) {
        v16bf a = frag_a(qb, INNER, qrow0, col0 + k0);
#pragma unroll
        for (int t = 0; t < 2; ++t) {
            int f0 = fblk * 64 + (ng * 2 + t) * 16;
            v16bf bb = frag_b(Kb, INNER, b * FQ + f0, col0 + k0);
            acc[t] = wmma_bf16(a, bb, acc[t]);
        }
    }

    int lane = threadIdx.x & 31;
    int lh = lane >> 4;
    int nl = lane & 15;
#pragma unroll
    for (int t = 0; t < 2; ++t) {
        int f = fblk * 64 + (ng * 2 + t) * 16 + nl;
        bool dead = (f < Fn) && (mask[b * Fn + f] == 0);
#pragma unroll
        for (int v = 0; v < 8; ++v) {
            int qq = qt * 16 + 8 * lh + v;
            float val = dead ? -__builtin_inff() : acc[t][v];
            sim[((size_t)bh * Qc + qq) * FQ + f] = val;
        }
    }
}

// ---------------------------------------------------------------------------
// Kernel 5: row softmax over FQ=4160, f32 logits -> bf16 probabilities.
// ---------------------------------------------------------------------------
__global__ __launch_bounds__(256)
void k_softmax(const float* __restrict__ sim, bf16* __restrict__ P) {
    int row = blockIdx.x;                  // B*H*Q = 8192
    const float* x = sim + (size_t)row * FQ;
    __shared__ float red[256];
    int t = threadIdx.x;

    float m = -__builtin_inff();
    for (int i = t; i < FQ; i += 256) m = fmaxf(m, x[i]);
    red[t] = m; __syncthreads();
    for (int o = 128; o > 0; o >>= 1) {
        if (t < o) red[t] = fmaxf(red[t], red[t + o]);
        __syncthreads();
    }
    m = red[0];
    __syncthreads();

    float s = 0.f;
    for (int i = t; i < FQ; i += 256) s += __expf(x[i] - m);
    red[t] = s; __syncthreads();
    for (int o = 128; o > 0; o >>= 1) {
        if (t < o) red[t] += red[t + o];
        __syncthreads();
    }
    float inv = 1.0f / red[0];

    bf16* p = P + (size_t)row * FQ;
    for (int i = t; i < FQ; i += 256) p[i] = (bf16)(__expf(x[i] - m) * inv);
}

// ---------------------------------------------------------------------------
// Kernel 6: out = P @ V  per (b,h): [64,4160] @ Vt[dh][4160]^T -> [64,64].
// Writes into attn[b*Q+q][h*64+dh] (bf16), ready for the Wo GEMM.
// ---------------------------------------------------------------------------
__global__ __launch_bounds__(256)
void k_pv(const bf16* __restrict__ P, const bf16* __restrict__ Vt,
          bf16* __restrict__ attn) {
    int bh = blockIdx.x;
    int b = bh >> 4, h = bh & 15;
    int wave = threadIdx.x >> 5;
    int qt = wave & 3;
    int ng = wave >> 2;

    const bf16* Pb = P  + (size_t)bh * Qc * FQ;
    const bf16* Vb = Vt + (size_t)bh * DHn * FQ;

    v8f acc[2] = {};
    for (int k0 = 0; k0 < FQ; k0 += 32) {
        v16bf a = frag_a(Pb, FQ, qt * 16, k0);
#pragma unroll
        for (int t = 0; t < 2; ++t) {
            v16bf bb = frag_b(Vb, FQ, (ng * 2 + t) * 16, k0);
            acc[t] = wmma_bf16(a, bb, acc[t]);
        }
    }

    int lane = threadIdx.x & 31;
    int lh = lane >> 4;
    int nl = lane & 15;
#pragma unroll
    for (int t = 0; t < 2; ++t) {
        int dh = (ng * 2 + t) * 16 + nl;
#pragma unroll
        for (int v = 0; v < 8; ++v) {
            int qq = qt * 16 + 8 * lh + v;
            attn[((size_t)(b * Qc + qq)) * INNER + h * DHn + dh] = (bf16)acc[t][v];
        }
    }
}

// ---------------------------------------------------------------------------
// Host launcher
// ---------------------------------------------------------------------------
extern "C" void kernel_launch(void* const* d_in, const int* in_sizes, int n_in,
                              void* d_out, int out_size, void* d_ws, size_t ws_size,
                              hipStream_t stream) {
    const float* features = (const float*)d_in[0];   // [8,4096,1024]
    const float* latents  = (const float*)d_in[1];   // [8,64,1024]
    const int*   kmask    = (const int*)  d_in[2];   // [8,4096]
    const float* ln_m_w   = (const float*)d_in[3];
    const float* ln_m_b   = (const float*)d_in[4];
    const float* ln_l_w   = (const float*)d_in[5];
    const float* ln_l_b   = (const float*)d_in[6];
    const float* Wq       = (const float*)d_in[7];
    const float* Wk       = (const float*)d_in[8];
    const float* Wv       = (const float*)d_in[9];
    const float* Wo       = (const float*)d_in[10];
    float* out = (float*)d_out;                      // [8,64,1024]

    // workspace carve-up
    char* ws = (char*)d_ws;
    size_t off = 0;
    auto carve = [&](size_t bytes) {
        void* p = ws + off;
        off += (bytes + 255) & ~(size_t)255;
        return p;
    };
    const size_t WB = (size_t)Dn * INNER * sizeof(bf16);       // 2 MB per weight
    bf16* Wqt   = (bf16*)carve(WB);
    bf16* Wkt   = (bf16*)carve(WB);
    bf16* Wvt   = (bf16*)carve(WB);
    bf16* Wot   = (bf16*)carve(WB);
    bf16* kv_ln = (bf16*)carve((size_t)Bn * FQ * Dn * sizeof(bf16));       // 65 MB
    bf16* latc  = (bf16*)carve((size_t)Bn * Qc * Dn * sizeof(bf16));       // 1 MB
    bf16* Kbuf  = (bf16*)carve((size_t)Bn * FQ * INNER * sizeof(bf16));    // 65 MB
    bf16* Vt    = (bf16*)carve((size_t)Bn * Hn * DHn * FQ * sizeof(bf16)); // 65 MB
    bf16* qbuf  = (bf16*)carve((size_t)Bn * Qc * INNER * sizeof(bf16));    // 1 MB
    float* sim  = (float*)carve((size_t)Bn * Hn * Qc * FQ * sizeof(float));// 130 MB
    bf16* Pbuf  = (bf16*)carve((size_t)Bn * Hn * Qc * FQ * sizeof(bf16));  // 65 MB
    bf16* attn  = (bf16*)carve((size_t)Bn * Qc * INNER * sizeof(bf16));    // 1 MB

    // 1) weights -> bf16, transposed to [N][K]
    int wgrid = (Dn * INNER) / 256;
    k_wconv<<<wgrid, 256, 0, stream>>>(Wq, Wqt);
    k_wconv<<<wgrid, 256, 0, stream>>>(Wk, Wkt);
    k_wconv<<<wgrid, 256, 0, stream>>>(Wv, Wvt);
    k_wconv<<<wgrid, 256, 0, stream>>>(Wo, Wot);

    // 2) layernorm -> bf16 (features into kv rows, latents into kv tail + compact)
    k_layernorm<<<Bn * Fn, 256, 0, stream>>>(features, ln_m_w, ln_m_b,
                                             kv_ln, nullptr, Fn, 0);
    k_layernorm<<<Bn * Qc, 256, 0, stream>>>(latents, ln_l_w, ln_l_b,
                                             kv_ln, latc, Qc, 1);

    // 3) projections (bf16 WMMA GEMMs)
    dim3 gKV((Bn * FQ) / 128, INNER / 128);           // 260 x 8
    k_gemm<<<gKV, 256, 0, stream>>>(kv_ln, Wkt, Kbuf, Dn, INNER, 1.0f, 0);
    k_gemm<<<gKV, 256, 0, stream>>>(kv_ln, Wvt, Vt,   Dn, INNER, 1.0f, 1);
    dim3 gQ((Bn * Qc) / 128, INNER / 128);            // 4 x 8
    const float scale = 0.125f;                       // DH^-0.5
    k_gemm<<<gQ, 256, 0, stream>>>(latc, Wqt, qbuf, Dn, INNER, scale, 0);

    // 4) masked logits
    dim3 gS(FQ / 64, Bn * Hn);                        // 65 x 128
    k_sim<<<gS, 256, 0, stream>>>(qbuf, Kbuf, kmask, sim);

    // 5) softmax -> bf16 probabilities
    k_softmax<<<Bn * Hn * Qc, 256, 0, stream>>>(sim, Pbuf);

    // 6) P @ V per head
    k_pv<<<Bn * Hn, 256, 0, stream>>>(Pbuf, Vt, attn);

    // 7) final projection -> f32 output
    k_gemm<<<gQ, 256, 0, stream>>>(attn, Wot, out, INNER, Dn, 1.0f, 2);

    (void)in_sizes; (void)n_in; (void)out_size; (void)ws_size;
}
